// LinearAttention_9294309228988
// MI455X (gfx1250) — matrix operand order repaired and checked by reading, not compiled, not run on hardware
//
#include <hip/hip_runtime.h>
#include <hip/hip_bf16.h>

typedef __attribute__((ext_vector_type(16))) __bf16 v16bf;
typedef __attribute__((ext_vector_type(8)))  __bf16 v8bf;
typedef __attribute__((ext_vector_type(8)))  float  v8f;
typedef __attribute__((ext_vector_type(4)))  int    v4i;
typedef __attribute__((address_space(1))) v4i v4i_g;   // global int4
typedef __attribute__((address_space(3))) v4i v4i_l;   // LDS int4

#define E_DIM 1024
#define SEQ   4096
#define BATCH 4
#define NHEAD 16
#define HDIM  64
#define MROWS (BATCH * SEQ)   // 16384
#define SCHUNK 512

#if defined(__HIP_DEVICE_COMPILE__) &&                                        \
    __has_builtin(__builtin_amdgcn_global_load_async_to_lds_b128) &&          \
    __has_builtin(__builtin_amdgcn_s_wait_asynccnt)
#define USE_ASYNC_LDS 1
#else
#define USE_ASYNC_LDS 0
#endif

__device__ __forceinline__ unsigned short f2bfu(float f) {
  return __builtin_bit_cast(unsigned short, (__bf16)f);
}
__device__ __forceinline__ float bfu2f(unsigned short u) {
  return (float)__builtin_bit_cast(__bf16, u);
}
__device__ __forceinline__ v16bf cat8(v8bf lo, v8bf hi) {
  return __builtin_shufflevector(lo, hi, 0, 1, 2, 3, 4, 5, 6, 7,
                                 8, 9, 10, 11, 12, 13, 14, 15);
}

// ---------------------------------------------------------------- cvt f32->bf16
__global__ __launch_bounds__(256)
void cvt_f32_bf16(const float* __restrict__ in, unsigned short* __restrict__ out,
                  int n) {
  int i = (blockIdx.x * 256 + threadIdx.x) * 4;
  if (i >= n) return;
  float4 f = *reinterpret_cast<const float4*>(in + i);
  unsigned long long p = (unsigned long long)f2bfu(f.x)
                       | ((unsigned long long)f2bfu(f.y) << 16)
                       | ((unsigned long long)f2bfu(f.z) << 32)
                       | ((unsigned long long)f2bfu(f.w) << 48);
  *reinterpret_cast<unsigned long long*>(out + i) = p;
}

__global__ __launch_bounds__(256)
void zero_f32(float* __restrict__ p, int n) {
  int i = blockIdx.x * 256 + threadIdx.x;
  if (i < n) p[i] = 0.f;
}

// ---------------------------------------------------------------- GEMM: C = act(A @ Bw^T + bias)
// A: MxK bf16 row-major, Bw: NxK bf16 row-major (torch Linear weight), bias: N f32.
// ACT: 0 = none, 1 = elu(x)+1. OUTF32: write f32 (else bf16 bits).
// Block = 256 threads = 8 waves; block tile 256(M) x 64(N); B tile (64x32)
// staged once per block in LDS via GLOBAL_LOAD_ASYNC_TO_LDS_B128; each wave
// computes a 32x64 strip: 2 A-fragments x 4 shared B-fragments = 8 WMMAs/k-step.
template <int ACT, bool OUTF32>
__global__ __launch_bounds__(256)
void gemm_bf16_wmma(const unsigned short* __restrict__ A,
                    const unsigned short* __restrict__ Bw,
                    const float* __restrict__ bias, void* __restrict__ Cout,
                    int M, int N, int K) {
  __shared__ alignas(16) unsigned short Bs[64][32];  // [n][k], 4 KB
  const int lane   = threadIdx.x & 31;
  const int wave   = threadIdx.x >> 5;
  const int hiHalf = lane >> 4;      // wave32: which K-half this lane holds
  const int kA     = hiHalf << 3;    // A frag: K base 0 or 8
  const int kB     = hiHalf << 4;    // 0 or 16; unused (B frag uses kA)
  const int nBase  = blockIdx.x * 64;
  const int mBase  = blockIdx.y * 256 + wave * 32;
  const unsigned short* arow0 = A + (size_t)(mBase + (lane & 15)) * K;
  const unsigned short* arow1 = arow0 + (size_t)16 * K;
  // staging assignment: thread -> (n, k-octet); one b128 per thread per k-step
  const int sn = threadIdx.x >> 2;         // 0..63
  const int sk = (threadIdx.x & 3) * 8;    // 0,8,16,24
  const unsigned short* srow = Bw + (size_t)(nBase + sn) * K + sk;

  v8f acc0[4] = {};
  v8f acc1[4] = {};
  for (int kk = 0; kk < K; kk += 32) {
#if USE_ASYNC_LDS
    __builtin_amdgcn_global_load_async_to_lds_b128(
        (v4i_g*)(srow + kk), (v4i_l*)&Bs[sn][sk], 0, 0);
    __builtin_amdgcn_s_wait_asynccnt(0);
#else
    *reinterpret_cast<v8bf*>(&Bs[sn][sk]) =
        *reinterpret_cast<const v8bf*>(srow + kk);
#endif
    __syncthreads();
    __builtin_prefetch(arow0 + kk + 256, 0, 0);
    // A 16x32 bf16 fragment: halfs[0..7]=K(kA..kA+7), halfs[8..15]=K(kA+16..kA+23)
    v16bf af0 = cat8(*reinterpret_cast<const v8bf*>(arow0 + kk + kA),
                     *reinterpret_cast<const v8bf*>(arow0 + kk + kA + 16));
    v16bf af1 = cat8(*reinterpret_cast<const v8bf*>(arow1 + kk + kA),
                     *reinterpret_cast<const v8bf*>(arow1 + kk + kA + 16));
#pragma unroll
    for (int nt = 0; nt < 4; ++nt) {
      // B 32x16 fragment: halfs[0..7]=K(kA..kA+7), halfs[8..15]=K(kA+16..kA+23)
      const unsigned short* bp = &Bs[nt * 16 + (lane & 15)][kA];
      v16bf bf = cat8(*reinterpret_cast<const v8bf*>(bp),
                      *reinterpret_cast<const v8bf*>(bp + 16));
      acc0[nt] = __builtin_amdgcn_wmma_f32_16x16x32_bf16(
          false, af0, false, bf, (short)0, acc0[nt], false, false);
      acc1[nt] = __builtin_amdgcn_wmma_f32_16x16x32_bf16(
          false, af1, false, bf, (short)0, acc1[nt], false, false);
    }
    __syncthreads();
  }
  // C/D layout: VGPR j -> M = j + 8*hiHalf, N = lane&15
  const int rowTop = mBase + (hiHalf << 3);
#pragma unroll
  for (int nt = 0; nt < 4; ++nt) {
    const int col = nBase + nt * 16 + (lane & 15);
    const float bv = bias[col];
#pragma unroll
    for (int j = 0; j < 8; ++j) {
      float v0 = acc0[nt][j] + bv;
      float v1 = acc1[nt][j] + bv;
      if (ACT == 1) {
        v0 = (v0 > 0.f) ? (v0 + 1.f) : __expf(v0);  // elu(x)+1
        v1 = (v1 > 0.f) ? (v1 + 1.f) : __expf(v1);
      }
      const size_t i0 = (size_t)(rowTop + j) * N + col;
      const size_t i1 = (size_t)(rowTop + 16 + j) * N + col;
      if (OUTF32) {
        reinterpret_cast<float*>(Cout)[i0] = v0;
        reinterpret_cast<float*>(Cout)[i1] = v1;
      } else {
        reinterpret_cast<unsigned short*>(Cout)[i0] = f2bfu(v0);
        reinterpret_cast<unsigned short*>(Cout)[i1] = f2bfu(v1);
      }
    }
  }
}

// ---------------------------------------------------------------- KV & ksum
// KV[bh][m][d] = sum_s v[b,s,h*64+m] * k[b,s,h*64+d];  Ksum[bh][d] = sum_s k.
// Block 256 thr = 8 waves; block handles one (b,h) and an s-chunk of 512.
// k/v tiles staged TRANSPOSED in LDS ([d][s]) so WMMA fragments are contiguous.
__global__ __launch_bounds__(256)
void kv_wmma(const unsigned short* __restrict__ Kb,
             const unsigned short* __restrict__ Vb, float* __restrict__ KV,
             float* __restrict__ Ksum) {
  __shared__ alignas(16) unsigned short kT[64][40];  // [d][s], pad 8 elems
  __shared__ alignas(16) unsigned short vT[64][40];  // [m][s]
  const int bh = blockIdx.x;
  const int b = bh >> 4, h = bh & 15;
  const int lane = threadIdx.x & 31, wave = threadIdx.x >> 5;
  const int hiHalf = lane >> 4;
  const int kA = hiHalf << 3;
  const int kB = hiHalf << 4;
  const int ls  = threadIdx.x >> 3;        // s row 0..31 loaded by this thread
  const int ld0 = (threadIdx.x & 7) * 8;   // d col base

  v8f acc[2] = {};
  float ks[8] = {};

  for (int it = 0; it < SCHUNK / 32; ++it) {
    const int sBase = blockIdx.y * SCHUNK + it * 32;
    {
      const size_t g = (size_t)(b * SEQ + sBase + ls) * E_DIM + h * HDIM + ld0;
      v8bf k8 = *reinterpret_cast<const v8bf*>(Kb + g);
      v8bf v8 = *reinterpret_cast<const v8bf*>(Vb + g);
#pragma unroll
      for (int j = 0; j < 8; ++j) {
        kT[ld0 + j][ls] = __builtin_bit_cast(unsigned short, k8[j]);
        vT[ld0 + j][ls] = __builtin_bit_cast(unsigned short, v8[j]);
        ks[j] += (float)k8[j];
      }
    }
    __syncthreads();
    // KV tile = A(v^T: [m][s]) x B(k: [s][d]); each wave does 2 of the 16 tiles
#pragma unroll
    for (int t = 0; t < 2; ++t) {
      const int tid = wave * 2 + t;
      const int mt = tid >> 2, dt = tid & 3;
      const unsigned short* vrow = &vT[mt * 16 + (lane & 15)][0];
      v16bf af = cat8(*reinterpret_cast<const v8bf*>(vrow + kA),
                      *reinterpret_cast<const v8bf*>(vrow + kA + 16));
      const unsigned short* krow = &kT[dt * 16 + (lane & 15)][0];
      v16bf bf = cat8(*reinterpret_cast<const v8bf*>(krow + kA),
                      *reinterpret_cast<const v8bf*>(krow + kA + 16));
      acc[t] = __builtin_amdgcn_wmma_f32_16x16x32_bf16(
          false, af, false, bf, (short)0, acc[t], false, false);
    }
    __syncthreads();
  }
#pragma unroll
  for (int j = 0; j < 8; ++j)
    atomicAdd(&Ksum[bh * HDIM + ld0 + j], ks[j]);
#pragma unroll
  for (int t = 0; t < 2; ++t) {
    const int tid = wave * 2 + t;
    const int mt = tid >> 2, dt = tid & 3;
    const int d = dt * 16 + (lane & 15);
    const int mTop = mt * 16 + (hiHalf << 3);
#pragma unroll
    for (int j = 0; j < 8; ++j)
      atomicAdd(&KV[((size_t)bh * HDIM + mTop + j) * HDIM + d], acc[t][j]);
  }
}

// ---------------------------------------------------------------- attn apply
// out[b,s,h*64+m] = Z(s) * sum_d q[b,s,h*64+d] * KV[bh][m][d]
// Block 128 thr = 4 waves; block -> one (b,h) x 64 s-rows; grid (64, 64).
__global__ __launch_bounds__(128)
void attn_wmma(const unsigned short* __restrict__ Qb,
               const float* __restrict__ KV, const float* __restrict__ Ksum,
               unsigned short* __restrict__ Attn) {
  __shared__ alignas(16) unsigned short kvbf[64][64];  // KV[m][d] as bf16
  __shared__ float zbuf[64];
  __shared__ float ksl[64];
  const int bh = blockIdx.x;
  const int b = bh >> 4, h = bh & 15;
  const int sBlk = blockIdx.y * 64;
  const int lane = threadIdx.x & 31, wave = threadIdx.x >> 5;
  const int hiHalf = lane >> 4;
  const int kA = hiHalf << 3;
  const int kB = hiHalf << 4;

  if (threadIdx.x < 64) ksl[threadIdx.x] = Ksum[bh * HDIM + threadIdx.x];
  for (int i = threadIdx.x; i < HDIM * HDIM; i += 128)
    kvbf[i >> 6][i & 63] = f2bfu(KV[(size_t)bh * HDIM * HDIM + i]);
  __syncthreads();
  if (threadIdx.x < 64) {
    const unsigned short* qr =
        Qb + (size_t)(b * SEQ + sBlk + threadIdx.x) * E_DIM + h * HDIM;
    float dot = 0.f;
    for (int d = 0; d < HDIM; ++d) dot += bfu2f(qr[d]) * ksl[d];
    zbuf[threadIdx.x] = 1.f / (dot + 1e-6f);
  }
  __syncthreads();

  const int row = sBlk + wave * 16 + (lane & 15);
  const unsigned short* qrow = Qb + (size_t)(b * SEQ + row) * E_DIM + h * HDIM;
  v8f acc[4] = {};
#pragma unroll
  for (int kk = 0; kk < HDIM; kk += 32) {
    v16bf af = cat8(*reinterpret_cast<const v8bf*>(qrow + kk + kA),
                    *reinterpret_cast<const v8bf*>(qrow + kk + kA + 16));
#pragma unroll
    for (int nt = 0; nt < 4; ++nt) {
      // B[d][m] = KV[m][d] -> lane-column m reads kvbf[m][kk+kA .. +7] and [kk+kA+16 .. +23]
      const unsigned short* kvr = &kvbf[nt * 16 + (lane & 15)][kk + kA];
      v16bf bf = cat8(*reinterpret_cast<const v8bf*>(kvr),
                      *reinterpret_cast<const v8bf*>(kvr + 16));
      acc[nt] = __builtin_amdgcn_wmma_f32_16x16x32_bf16(
          false, af, false, bf, (short)0, acc[nt], false, false);
    }
  }
  const int sTop = wave * 16 + (hiHalf << 3);
#pragma unroll
  for (int nt = 0; nt < 4; ++nt) {
    const int m = nt * 16 + (lane & 15);
#pragma unroll
    for (int j = 0; j < 8; ++j) {
      const int sl = sTop + j;
      const float v = acc[nt][j] * zbuf[sl];
      Attn[(size_t)(b * SEQ + sBlk + sl) * E_DIM + h * HDIM + m] = f2bfu(v);
    }
  }
}

// ---------------------------------------------------------------- launch
extern "C" void kernel_launch(void* const* d_in, const int* in_sizes, int n_in,
                              void* d_out, int out_size, void* d_ws,
                              size_t ws_size, hipStream_t stream) {
  (void)in_sizes; (void)n_in; (void)out_size; (void)ws_size;
  const float* hs = (const float*)d_in[0];
  const float* Wq = (const float*)d_in[1];
  const float* bq = (const float*)d_in[2];
  const float* Wk = (const float*)d_in[3];
  const float* bk = (const float*)d_in[4];
  const float* Wv = (const float*)d_in[5];
  const float* bv = (const float*)d_in[6];
  const float* Wo = (const float*)d_in[7];
  const float* bo = (const float*)d_in[8];
  float* out = (float*)d_out;

  char* ws = (char*)d_ws;
  size_t off = 0;
  auto take = [&](size_t bytes) -> char* {
    char* p = ws + off;
    off += (bytes + 255) & ~(size_t)255;
    return p;
  };
  const size_t nX = (size_t)MROWS * E_DIM;  // 16M elems
  const size_t nW = (size_t)E_DIM * E_DIM;  // 1M elems
  unsigned short* Xb  = (unsigned short*)take(nX * 2);
  unsigned short* Wqb = (unsigned short*)take(nW * 2);
  unsigned short* Wkb = (unsigned short*)take(nW * 2);
  unsigned short* Wvb = (unsigned short*)take(nW * 2);
  unsigned short* Wob = (unsigned short*)take(nW * 2);
  unsigned short* Qb  = (unsigned short*)take(nX * 2);
  unsigned short* Kb  = (unsigned short*)take(nX * 2);
  unsigned short* Vb  = (unsigned short*)take(nX * 2);
  unsigned short* Ab  = (unsigned short*)take(nX * 2);
  float* KVp   = (float*)take((size_t)64 * HDIM * HDIM * 4);  // 1 MB, 256-aligned
  float* Ksump = (float*)take((size_t)64 * HDIM * 4);         // contiguous after KVp

  cvt_f32_bf16<<<(int)(nX / 1024), 256, 0, stream>>>(hs, Xb, (int)nX);
  cvt_f32_bf16<<<(int)(nW / 1024), 256, 0, stream>>>(Wq, Wqb, (int)nW);
  cvt_f32_bf16<<<(int)(nW / 1024), 256, 0, stream>>>(Wk, Wkb, (int)nW);
  cvt_f32_bf16<<<(int)(nW / 1024), 256, 0, stream>>>(Wv, Wvb, (int)nW);
  cvt_f32_bf16<<<(int)(nW / 1024), 256, 0, stream>>>(Wo, Wob, (int)nW);

  dim3 ggrid(E_DIM / 64, MROWS / 256);
  gemm_bf16_wmma<1, false><<<ggrid, 256, 0, stream>>>(Xb, Wqb, bq, Qb, MROWS, E_DIM, E_DIM);
  gemm_bf16_wmma<1, false><<<ggrid, 256, 0, stream>>>(Xb, Wkb, bk, Kb, MROWS, E_DIM, E_DIM);
  gemm_bf16_wmma<0, false><<<ggrid, 256, 0, stream>>>(Xb, Wvb, bv, Vb, MROWS, E_DIM, E_DIM);

  const int nZero = 64 * HDIM * HDIM + 64 * HDIM;  // KV + Ksum (contiguous)
  zero_f32<<<(nZero + 255) / 256, 256, 0, stream>>>(KVp, nZero);

  kv_wmma<<<dim3(64, SEQ / SCHUNK), 256, 0, stream>>>(Kb, Vb, KVp, Ksump);
  attn_wmma<<<dim3(64, SEQ / 64), 128, 0, stream>>>(Qb, KVp, Ksump, Ab);

  gemm_bf16_wmma<0, true><<<ggrid, 256, 0, stream>>>(Ab, Wob, bo, out, MROWS, E_DIM, E_DIM);
}
